// PRISMModel_17978733101174
// MI455X (gfx1250) — matrix operand order repaired and checked by reading, not compiled, not run on hardware
//
#include <hip/hip_runtime.h>
#include <hip/hip_bf16.h>

// ---------------- model constants ----------------
constexpr int NLc = 64, GDc = 2, FDc = 8, NFc = 65;
constexpr int Bc = 512, Tc = 64, INc = 128, SDc = 16, HIDc = 32, KCc = 10;
constexpr int TNc = Tc * NFc;             // 4160 flattened (t,f) columns
constexpr int FEATC = Bc * Tc * NFc * FDc; // 17,039,360 elements
constexpr int NALL = Bc + KCc;            // 522 graph nodes

typedef __attribute__((ext_vector_type(16))) _Float16 v16h;
typedef __attribute__((ext_vector_type(8)))  float    v8f;
typedef __attribute__((ext_vector_type(4)))  unsigned int u32x4;
typedef __attribute__((ext_vector_type(4)))  int      i32x4;
typedef __attribute__((ext_vector_type(8)))  int      i32x8;

__device__ __forceinline__ float sigm(float x) { return 1.f / (1.f + __expf(-x)); }

__device__ __forceinline__ v8f wmma_f16(v16h a, v16h b, v8f c) {
  // D = A(16x32 f16) * B(32x16 f16) + C(16x16 f32)
  return __builtin_amdgcn_wmma_f32_16x16x32_f16(false, a, false, b, (short)0, c, false, false);
}

// ---------------- K1: 64 micro-GRUs (GD=2 -> FD=8), per-(lane,batch) thread ----------------
__global__ void __launch_bounds__(128) k_micro_gru(
    const float* __restrict__ x, const float* __restrict__ w_ih,
    const float* __restrict__ w_hh, const float* __restrict__ b_ih,
    const float* __restrict__ b_hh, _Float16* __restrict__ featH) {
  __shared__ float sWih[24 * GDc], sWhh[24 * FDc], sBih[24], sBhh[24];
  const int lane = blockIdx.x;   // micro-cell 0..63
  const int tid = threadIdx.x;
  for (int i = tid; i < 24 * GDc; i += 128) sWih[i] = w_ih[lane * 24 * GDc + i];
  for (int i = tid; i < 24 * FDc; i += 128) sWhh[i] = w_hh[lane * 24 * FDc + i];
  if (tid < 24) { sBih[tid] = b_ih[lane * 24 + tid]; sBhh[tid] = b_hh[lane * 24 + tid]; }
  __syncthreads();
  const int b = blockIdx.y * 128 + tid;
  float h[FDc];
#pragma unroll
  for (int c = 0; c < FDc; c++) h[c] = 0.f;
  for (int t = 0; t < Tc; t++) {
    const float x0 = x[(b * Tc + t) * INc + lane * GDc + 0];
    const float x1 = x[(b * Tc + t) * INc + lane * GDc + 1];
    float gi[24], gh[24];
#pragma unroll
    for (int g = 0; g < 24; g++) {
      gi[g] = sWih[g * 2] * x0 + sWih[g * 2 + 1] * x1 + sBih[g];
      float a = sBhh[g];
#pragma unroll
      for (int d = 0; d < FDc; d++) a += sWhh[g * FDc + d] * h[d];
      gh[g] = a;
    }
#pragma unroll
    for (int c = 0; c < FDc; c++) {
      float r = sigm(gi[c] + gh[c]);
      float z = sigm(gi[8 + c] + gh[8 + c]);
      float n = tanhf(gi[16 + c] + r * gh[16 + c]);
      h[c] = (1.f - z) * n + z * h[c];
    }
    _Float16* dst = featH + ((size_t)(b * Tc + t) * NFc + lane) * FDc;
#pragma unroll
    for (int c = 0; c < FDc; c++) dst[c] = (_Float16)h[c];
  }
}

// ---------------- K2: static projection -> feature slot NF-1, broadcast over T ----------------
__global__ void k_static_proj(const float* __restrict__ st, const float* __restrict__ st_w,
                              const float* __restrict__ st_b, _Float16* __restrict__ featH) {
  int idx = blockIdx.x * blockDim.x + threadIdx.x;
  if (idx >= Bc * FDc) return;
  int b = idx >> 3, c = idx & 7;
  float a = st_b[c];
  for (int d = 0; d < SDc; d++) a += st[b * SDc + d] * st_w[c * SDc + d];
  _Float16 hv = (_Float16)a;
  for (int t = 0; t < Tc; t++)
    featH[((size_t)(b * Tc + t) * NFc + (NFc - 1)) * FDc + c] = hv;
}

// ---------------- K3: fused per-batch attention (TDM load, LDS-resident, WMMA P^T@Vsum) ----------------
__global__ void __launch_bounds__(256) k_attention(
    const _Float16* __restrict__ featH, const float* __restrict__ td,
    const float* __restrict__ wq, const float* __restrict__ bq,
    const float* __restrict__ wk, const float* __restrict__ bk,
    const float* __restrict__ wv, const float* __restrict__ bv,
    const float* __restrict__ decay, const float* __restrict__ gfe,
    const float* __restrict__ gm_p, const float* __restrict__ conf_p,
    _Float16* __restrict__ attnH, float* __restrict__ x_all, float* __restrict__ dc_all) {
  __shared__ _Float16 sFK[TNc * FDc];   // 66560 B: feat slice, then K-matrix in place
  __shared__ float sQ[NFc * FDc];       // queries (last timestep)
  __shared__ float sVs[NFc * FDc];      // Vsum (65x8)
  __shared__ float sSumV[FDc];
  __shared__ float sMax[NFc], sSum[NFc];
  __shared__ float sDcl[TNc];           // featsum scratch, then dcl[j]
  __shared__ float sDcp[NFc];
  __shared__ float sW[3 * 64 + 3 * 8];  // wq|wk|wv|bq|bk|bv
  const int tid = threadIdx.x;
  const int b = blockIdx.x;
  if (tid < 64) { sW[tid] = wq[tid]; sW[64 + tid] = wk[tid]; sW[128 + tid] = wv[tid]; }
  if (tid >= 64 && tid < 72) {
    int c = tid - 64; sW[192 + c] = bq[c]; sW[200 + c] = bk[c]; sW[208 + c] = bv[c];
  }
  if (tid < NFc) sDcp[tid] = 0.f;

  // ---- stage the 66.5KB feat slice into LDS via the Tensor Data Mover ----
#if __has_builtin(__builtin_amdgcn_tensor_load_to_lds)
  if (tid < 32) {  // wave 0 issues a single TDM descriptor
    const _Float16* src = featH + (size_t)b * (TNc * FDc);
    unsigned long long ga = (unsigned long long)(uintptr_t)src;
    unsigned int ldsAddr = (unsigned int)(uintptr_t)(&sFK[0]); // flat low 32 bits == LDS offset
    u32x4 g0; i32x8 g1; i32x4 g2, g3;
    g0[0] = 1u;                                   // count=1, user descriptor
    g0[1] = ldsAddr;                              // lds_addr
    g0[2] = (unsigned int)ga;                     // global_addr[31:0]
    g0[3] = (unsigned int)((ga >> 32) & 0x01FFFFFFull) | 0x80000000u; // addr[56:32] | type=2
    g1[0] = 0x00010000;                           // wg_mask=0, data_size=1 (2B)
    g1[1] = (int)0x82000000u;                     // tensor_dim0=33280 (low16 -> [31:16])
    g1[2] = 0x00010000;                           // tensor_dim0 hi=0, tensor_dim1=1
    g1[3] = (int)0x82000000u;                     // tensor_dim1 hi=0, tile_dim0=33280
    g1[4] = 1;                                    // tile_dim1=1, tile_dim2=0
    g1[5] = 33280;                                // tensor_dim0_stride low32
    g1[6] = 0; g1[7] = 0;
    g2 = (i32x4)0; g3 = (i32x4)0;
#if defined(__clang_major__) && __clang_major__ >= 23
    i32x8 g4 = (i32x8)0;
    __builtin_amdgcn_tensor_load_to_lds(g0, g1, g2, g3, g4, 0);
#else
    __builtin_amdgcn_tensor_load_to_lds(g0, g1, g2, g3, 0);
#endif
#if __has_builtin(__builtin_amdgcn_s_wait_tensorcnt)
    __builtin_amdgcn_s_wait_tensorcnt(0);
#endif
  }
#else
  for (int i = tid; i < TNc * FDc; i += 256) sFK[i] = featH[(size_t)b * TNc * FDc + i];
#endif
  __syncthreads();

  // ---- featsum (scratch in sDcl) and queries ----
  for (int idx = tid; idx < NFc * FDc; idx += 256) {
    int f = idx >> 3, d = idx & 7;
    float a = 0.f;
    for (int t = 0; t < Tc; t++) a += (float)sFK[(t * NFc + f) * FDc + d];
    sDcl[idx] = a;
  }
  for (int idx = tid; idx < NFc * FDc; idx += 256) {
    int q = idx >> 3, c = idx & 7;
    float a = sW[192 + c];
    const _Float16* fr = &sFK[((Tc - 1) * NFc + q) * FDc];
#pragma unroll
    for (int d = 0; d < FDc; d++) a += (float)fr[d] * sW[c * FDc + d];
    sQ[idx] = a;
  }
  __syncthreads();
  // ---- Vsum = featsum @ wv^T + T*bv ; x_avg ----
  for (int idx = tid; idx < NFc * FDc; idx += 256) {
    int f = idx >> 3, c = idx & 7;
    float a = (float)Tc * sW[208 + c];
#pragma unroll
    for (int d = 0; d < FDc; d++) a += sDcl[f * FDc + d] * sW[128 + c * FDc + d];
    sVs[idx] = a;
  }
  if (tid < NFc) {
    float a = 0.f;
#pragma unroll
    for (int d = 0; d < FDc; d++) a += sDcl[tid * FDc + d];
    x_all[b * NFc + tid] = a / (float)(Tc * FDc);
  }
  __syncthreads();
  if (tid < FDc) {
    float a = 0.f;
    for (int f = 0; f < NFc; f++) a += sVs[f * FDc + tid];
    sSumV[tid] = a;
  }
  // ---- in-place K = feat @ wk^T + bk ----
  for (int j = tid; j < TNc; j += 256) {
    float in[FDc];
#pragma unroll
    for (int d = 0; d < FDc; d++) in[d] = (float)sFK[j * FDc + d];
#pragma unroll
    for (int c = 0; c < FDc; c++) {
      float a = sW[200 + c];
#pragma unroll
      for (int d = 0; d < FDc; d++) a += in[d] * sW[64 + c * FDc + d];
      sFK[j * FDc + c] = (_Float16)a;
    }
  }
  __syncthreads();

  // ---- softmax row stats: each wave owns q = wv_, wv_+8, ... ----
  const int wv_ = tid >> 5, ln = tid & 31;
  const float rs8 = 0.35355339059327373f;  // 1/sqrt(8)
  for (int q = wv_; q < NFc; q += 8) {
    float m = -3.4e38f;
    for (int j = ln; j < TNc; j += 32) {
      float s = 0.f;
#pragma unroll
      for (int d = 0; d < FDc; d++) s += sQ[q * FDc + d] * (float)sFK[j * FDc + d];
      m = fmaxf(m, s * rs8);
    }
#pragma unroll
    for (int off = 16; off; off >>= 1) m = fmaxf(m, __shfl_xor(m, off, 32));
    float ss = 0.f;
    for (int j = ln; j < TNc; j += 32) {
      float s = 0.f;
#pragma unroll
      for (int d = 0; d < FDc; d++) s += sQ[q * FDc + d] * (float)sFK[j * FDc + d];
      ss += __expf(s * rs8 - m);
    }
#pragma unroll
    for (int off = 16; off; off >>= 1) ss += __shfl_xor(ss, off, 32);
    if (ln == 0) { sMax[q] = m; sSum[q] = ss; }
  }
  __syncthreads();

  // ---- per-column: am -> dcl, dc_pat ----
  const float gmv = gm_p[0], cp = conf_p[0];
  for (int j = tid; j < TNc; j += 256) {
    float kr[FDc];
#pragma unroll
    for (int d = 0; d < FDc; d++) kr[d] = (float)sFK[j * FDc + d];
    float am = 0.f;
    for (int q = 0; q < NFc; q++) {
      float s = 0.f;
#pragma unroll
      for (int d = 0; d < FDc; d++) s += sQ[q * FDc + d] * kr[d];
      am += __expf(s * rs8 - sMax[q]) / sSum[q];
    }
    am *= (1.f / NFc);
    int f = j % NFc;
    float tdc = fminf(td[(size_t)b * TNc + j], 2.0f);
    float divide = decay[f] * logf(2.718281828459045f + (1.f - am) * tdc);
    float dcl = tanhf(am / divide);
    if (tdc >= 2.0f) dcl = gmv * gfe[f];
    sDcl[j] = dcl;
    atomicAdd(&sDcp[f], dcl * (1.f / Tc));
  }
  __syncthreads();
  if (tid < NFc) dc_all[b * NFc + tid] = sDcp[tid];

  // ---- WMMA: attn_out = (1-cp) * P^T @ Vsum + cp * dcl (x) sumV ----
  const int lrow = ln & 15, hf = ln >> 4;
  v16h bf[3];
#pragma unroll
  for (int ks = 0; ks < 3; ks++) {
#pragma unroll
    for (int e = 0; e < 16; e++) {
      int kq = ks * 32 + ((e < 8) ? (e + 8 * hf) : (e + 8 + 8 * hf));      // B layout: K = table order (matches A), N = lane%16
      float v = (kq < NFc && lrow < FDc) ? sVs[kq * FDc + lrow] : 0.f;
      bf[ks][e] = (_Float16)v;
    }
  }
  const float omc = 1.f - cp;
  for (int jt = wv_; jt < TNc / 16; jt += 8) {
    const int j0 = jt * 16;
    const int jj = j0 + lrow;             // A layout: M = lane%16
    float kr[FDc];
#pragma unroll
    for (int d = 0; d < FDc; d++) kr[d] = (float)sFK[jj * FDc + d];
    v8f acc = {};
#pragma unroll
    for (int ks = 0; ks < 3; ks++) {
      v16h af;
#pragma unroll
      for (int e = 0; e < 16; e++) {
        int k = (e < 8) ? (e + 8 * hf) : (e + 8 + 8 * hf);  // A layout
        int q = ks * 32 + k;
        float p = 0.f;
        if (q < NFc) {
          float s = 0.f;
#pragma unroll
          for (int d = 0; d < FDc; d++) s += sQ[q * FDc + d] * kr[d];
          p = __expf(s * rs8 - sMax[q]) / sSum[q];
        }
        af[e] = (_Float16)p;
      }
      acc = wmma_f16(af, bf[ks], acc);
    }
    if (lrow < FDc) {
#pragma unroll
      for (int vv = 0; vv < 8; vv++) {
        int jr = j0 + vv + 8 * hf;        // D layout: M = v + 8*half
        float o = omc * acc[vv] + cp * sDcl[jr] * sSumV[lrow];
        attnH[(size_t)b * (TNc * FDc) + jr * FDc + lrow] = (_Float16)o;
      }
    }
  }
}

// ---------------- K4..K7: graph construction (small, VALU) ----------------
__global__ void k_centers(const float* __restrict__ rc, const float* __restrict__ cp_w,
                          const float* __restrict__ cp_b, const float* __restrict__ cconf,
                          float* __restrict__ x_all, float* __restrict__ dc_all) {
  int idx = blockIdx.x * blockDim.x + threadIdx.x;
  if (idx >= KCc * NFc) return;
  int k = idx / NFc, f = idx - k * NFc;
  float a = cp_b[f];
  for (int d = 0; d < INc; d++) a += rc[k * INc + d] * cp_w[f * INc + d];
  x_all[(Bc + k) * NFc + f] = a;
  dc_all[(Bc + k) * NFc + f] = cconf[k * NFc + f];
}

__global__ void k_prep(const float* __restrict__ x_all, const float* __restrict__ dc_all,
                       float* __restrict__ sq, float* __restrict__ aexp) {
  int i = blockIdx.x * blockDim.x + threadIdx.x;
  if (i >= NALL) return;
  float s = 0.f;
  for (int f = 0; f < NFc; f++) {
    float v = x_all[i * NFc + f];
    s += v * v;
    aexp[i * NFc + f] = __expf(1.f - dc_all[i * NFc + f]);
  }
  sq[i] = s;
}

__global__ void k_adj(const float* __restrict__ x_all, const float* __restrict__ aexp,
                      const float* __restrict__ sq, const float* __restrict__ adj_p,
                      float* __restrict__ adj) {
  int idx = blockIdx.x * blockDim.x + threadIdx.x;
  if (idx >= NALL * NALL) return;
  int i = idx / NALL, j = idx - i * NALL;
  float dot = 0.f, cf = 0.f;
  for (int f = 0; f < NFc; f++) {
    dot += x_all[i * NFc + f] * x_all[j * NFc + f];
    cf += aexp[i * NFc + f] * aexp[j * NFc + f];
  }
  float msd = (sq[i] + sq[j] - 2.f * dot) / (float)NFc;
  cf /= (float)NFc;
  float pA = adj_p[0];
  float sim = 1.f / ((1.f - pA) * msd + pA * cf);
  adj[idx] = (i == j) ? 1.f : sim;
}

__global__ void k_glabel(const float* __restrict__ adj, int* __restrict__ glabel) {
  int i = blockIdx.x * blockDim.x + threadIdx.x;
  if (i >= Bc) return;
  float best = adj[i * NALL + Bc];
  int bi = 0;
  for (int k = 1; k < KCc; k++) {
    float v = adj[i * NALL + Bc + k];
    if (v > best) { best = v; bi = k; }
  }
  glabel[i] = bi;
}

__global__ void k_xg(const float* __restrict__ x_all, const float* __restrict__ gcn_w,
                     float* __restrict__ xg) {
  int idx = blockIdx.x * blockDim.x + threadIdx.x;
  if (idx >= NALL * HIDc) return;
  int i = idx / HIDc, h = idx - i * HIDc;
  float a = 0.f;
  for (int f = 0; f < NFc; f++) a += x_all[i * NFc + f] * gcn_w[f * HIDc + h];
  xg[idx] = a;
}

__global__ void __launch_bounds__(256) k_graph_small(
    const float* __restrict__ adj, const float* __restrict__ xg,
    const float* __restrict__ gcn_b, const float* __restrict__ eo_w,
    const float* __restrict__ eo_b, const float* __restrict__ imp1_w,
    const float* __restrict__ imp1_b, const float* __restrict__ imp2_w,
    const float* __restrict__ imp2_b, float* __restrict__ gemb, float* __restrict__ gsc) {
  __shared__ float agg[KCc * HIDc];
  __shared__ float ge[KCc * NFc];
  __shared__ float t16[KCc * 16];
  const int tid = threadIdx.x;
  for (int idx = tid; idx < KCc * HIDc; idx += 256) {
    int k = idx / HIDc, h = idx - k * HIDc;
    float a = gcn_b[h];
    for (int j = 0; j < NALL; j++) a += adj[(Bc + k) * NALL + j] * xg[j * HIDc + h];
    agg[idx] = a;
  }
  __syncthreads();
  for (int idx = tid; idx < KCc * NFc; idx += 256) {
    int k = idx / NFc, f = idx - k * NFc;
    float a = eo_b[f];
    for (int h = 0; h < HIDc; h++) a += agg[k * HIDc + h] * eo_w[f * HIDc + h];
    ge[idx] = sigm(a);
  }
  __syncthreads();
  for (int idx = tid; idx < KCc * 16; idx += 256) {
    int k = idx / 16, m = idx - k * 16;
    float a = imp1_b[m];
    for (int f = 0; f < NFc; f++) a += ge[k * NFc + f] * imp1_w[m * NFc + f];
    t16[idx] = 0.5f * a * (1.f + erff(a * 0.7071067811865476f));  // exact gelu
  }
  __syncthreads();
  for (int idx = tid; idx < KCc * NFc; idx += 256) {
    int k = idx / NFc, f = idx - k * NFc;
    float a = imp2_b[f];
    for (int m = 0; m < 16; m++) a += t16[k * 16 + m] * imp2_w[f * 16 + m];
    gsc[idx] = sigm(a);
    gemb[idx] = ge[idx];
  }
}

// ---------------- K8a: WMMA GEMM (BT x 520) @ (520 x 65): h*gscores @ p1 + ge_p*gpe ----------------
__global__ void __launch_bounds__(256) k_p1(
    const _Float16* __restrict__ attnH, const float* __restrict__ p1_w,
    const float* __restrict__ p1_b, const float* __restrict__ gsc,
    const float* __restrict__ gemb, const int* __restrict__ glabel,
    const float* __restrict__ ge_p, _Float16* __restrict__ Hout) {
  const int tid = threadIdx.x, ln = tid & 31, wv_ = tid >> 5;
  const int widx = blockIdx.x * 8 + wv_;     // one 16x16 tile per wave; 2048x5 tiles exactly
  const int mt = widx / 5, nt = widx - mt * 5;
  const int lrow = ln & 15, hf = ln >> 4;
  const int r = mt * 16 + lrow;              // output row = b*T + t
  const int gl = glabel[r / Tc];
  const float* gr = gsc + gl * NFc;
  const int n = nt * 16 + lrow;              // output column (feature f')
  if (n < NFc) __builtin_prefetch(p1_w + n * 520, 0, 1);
  v8f acc = {};
  for (int ks = 0; ks < 17; ks++) {          // K = 520 padded to 544
    v16h af, bf;
#pragma unroll
    for (int e = 0; e < 16; e++) {
      int k = (e < 8) ? (e + 8 * hf) : (e + 8 + 8 * hf);
      int kk = ks * 32 + k;
      float av = 0.f;
      if (kk < 520) av = (float)attnH[(size_t)r * 520 + kk] * gr[kk >> 3];
      af[e] = (_Float16)av;
      int kb = ks * 32 + k;
      float bv = (kb < 520 && n < NFc) ? p1_w[n * 520 + kb] : 0.f;
      bf[e] = (_Float16)bv;
    }
    acc = wmma_f16(af, bf, acc);
  }
  const float gep = ge_p[0];
  if (n < NFc) {
#pragma unroll
    for (int vv = 0; vv < 8; vv++) {
      int rr = mt * 16 + vv + 8 * hf;
      int gl2 = glabel[rr / Tc];
      float o = acc[vv] + p1_b[n] + gep * gemb[gl2 * NFc + n];
      Hout[(size_t)rr * NFc + n] = (_Float16)o;
    }
  }
}

// ---------------- K8b: WMMA GEMM (BT x 65) @ (65 x 32): p2 projection ----------------
__global__ void __launch_bounds__(256) k_p2(
    const _Float16* __restrict__ Hout, const float* __restrict__ p2_w,
    const float* __restrict__ p2_b, float* __restrict__ seqb) {
  const int tid = threadIdx.x, ln = tid & 31, wv_ = tid >> 5;
  const int widx = blockIdx.x * 8 + wv_;     // 2048x2 tiles exactly
  const int mt = widx >> 1, nt = widx & 1;
  const int lrow = ln & 15, hf = ln >> 4;
  const int r = mt * 16 + lrow;
  const int n = nt * 16 + lrow;              // < 32 always
  v8f acc = {};
  for (int ks = 0; ks < 3; ks++) {           // K = 65 padded to 96
    v16h af, bf;
#pragma unroll
    for (int e = 0; e < 16; e++) {
      int k = (e < 8) ? (e + 8 * hf) : (e + 8 + 8 * hf);
      int kk = ks * 32 + k;
      af[e] = (_Float16)((kk < NFc) ? (float)Hout[(size_t)r * NFc + kk] : 0.f);
      int kb = ks * 32 + k;
      bf[e] = (_Float16)((kb < NFc) ? p2_w[n * NFc + kb] : 0.f);
    }
    acc = wmma_f16(af, bf, acc);
  }
#pragma unroll
  for (int vv = 0; vv < 8; vv++) {
    int rr = mt * 16 + vv + 8 * hf;
    seqb[(size_t)rr * HIDc + n] = acc[vv] + p2_b[n];
  }
}

// ---------------- K9: 2-layer GRU (HID=32) per-batch thread + head ----------------
__global__ void __launch_bounds__(256) k_gru2(
    const float* __restrict__ seqb, const float* __restrict__ w_ih,
    const float* __restrict__ w_hh, const float* __restrict__ b_ih,
    const float* __restrict__ b_hh, const float* __restrict__ head_w,
    const float* __restrict__ head_b, float* __restrict__ y) {
  __shared__ float sW[12672];  // [wih(2*3072) | whh(2*3072) | bih(2*96) | bhh(2*96)]
  const int tid = threadIdx.x;
  for (int i = tid; i < 2 * 96 * 32; i += 256) { sW[i] = w_ih[i]; sW[6144 + i] = w_hh[i]; }
  for (int i = tid; i < 2 * 96; i += 256) { sW[12288 + i] = b_ih[i]; sW[12480 + i] = b_hh[i]; }
  __syncthreads();
  const int b = blockIdx.x * 256 + tid;
  float h1[HIDc], h2[HIDc], xin[HIDc], hn[HIDc];
  for (int c = 0; c < HIDc; c++) { h1[c] = 0.f; h2[c] = 0.f; }
  for (int t = 0; t < Tc; t++) {
    for (int d = 0; d < HIDc; d++) xin[d] = seqb[(size_t)(b * Tc + t) * HIDc + d];
    for (int l = 0; l < 2; l++) {
      const float* wih = sW + l * 3072;
      const float* whh = sW + 6144 + l * 3072;
      const float* bih = sW + 12288 + l * 96;
      const float* bhh = sW + 12480 + l * 96;
      float* hp = (l == 0) ? h1 : h2;
      const float* in = (l == 0) ? xin : h1;
      for (int c = 0; c < HIDc; c++) {
        float gir = bih[c], giz = bih[32 + c], gin = bih[64 + c];
        float ghr = bhh[c], ghz = bhh[32 + c], ghn = bhh[64 + c];
        for (int d = 0; d < HIDc; d++) {
          float xd = in[d], hd = hp[d];
          gir += wih[c * 32 + d] * xd; giz += wih[(32 + c) * 32 + d] * xd; gin += wih[(64 + c) * 32 + d] * xd;
          ghr += whh[c * 32 + d] * hd; ghz += whh[(32 + c) * 32 + d] * hd; ghn += whh[(64 + c) * 32 + d] * hd;
        }
        float rr = sigm(gir + ghr), zz = sigm(giz + ghz);
        float nn = tanhf(gin + rr * ghn);
        hn[c] = (1.f - zz) * nn + zz * hp[c];
      }
      for (int c = 0; c < HIDc; c++) hp[c] = hn[c];
    }
  }
  float a = head_b[0];
  for (int d = 0; d < HIDc; d++) a += h2[d] * head_w[d];
  y[b] = a;
}

// ---------------- launcher ----------------
extern "C" void kernel_launch(void* const* d_in, const int* in_sizes, int n_in,
                              void* d_out, int out_size, void* d_ws, size_t ws_size,
                              hipStream_t stream) {
  (void)in_sizes; (void)n_in; (void)out_size; (void)ws_size;
  const float* x        = (const float*)d_in[0];
  const float* td       = (const float*)d_in[1];
  const float* st       = (const float*)d_in[2];
  const float* gfe      = (const float*)d_in[3];
  const float* mc_w_ih  = (const float*)d_in[4];
  const float* mc_w_hh  = (const float*)d_in[5];
  const float* mc_b_ih  = (const float*)d_in[6];
  const float* mc_b_hh  = (const float*)d_in[7];
  const float* st_w     = (const float*)d_in[8];
  const float* st_b     = (const float*)d_in[9];
  const float* wq       = (const float*)d_in[10];
  const float* bq       = (const float*)d_in[11];
  const float* wk       = (const float*)d_in[12];
  const float* bk       = (const float*)d_in[13];
  const float* wv       = (const float*)d_in[14];
  const float* bv       = (const float*)d_in[15];
  const float* decay    = (const float*)d_in[16];
  const float* gm       = (const float*)d_in[17];
  const float* conf_p   = (const float*)d_in[18];
  const float* raw_c    = (const float*)d_in[19];
  const float* cp_w     = (const float*)d_in[20];
  const float* cp_b     = (const float*)d_in[21];
  const float* adj_p    = (const float*)d_in[22];
  const float* gcn_w    = (const float*)d_in[23];
  const float* gcn_b    = (const float*)d_in[24];
  const float* eo_w     = (const float*)d_in[25];
  const float* eo_b     = (const float*)d_in[26];
  const float* cconf    = (const float*)d_in[27];
  const float* imp1_w   = (const float*)d_in[28];
  const float* imp1_b   = (const float*)d_in[29];
  const float* imp2_w   = (const float*)d_in[30];
  const float* imp2_b   = (const float*)d_in[31];
  const float* ge_p     = (const float*)d_in[32];
  const float* p1_w     = (const float*)d_in[33];
  const float* p1_b     = (const float*)d_in[34];
  const float* p2_w     = (const float*)d_in[35];
  const float* p2_b     = (const float*)d_in[36];
  const float* gru_w_ih = (const float*)d_in[37];
  const float* gru_w_hh = (const float*)d_in[38];
  const float* gru_b_ih = (const float*)d_in[39];
  const float* gru_b_hh = (const float*)d_in[40];
  const float* head_w   = (const float*)d_in[41];
  const float* head_b   = (const float*)d_in[42];

  char* w = (char*)d_ws;
  auto alloc = [&](size_t bytes) -> char* {
    char* p = w; w += (bytes + 255) & ~(size_t)255; return p;
  };
  _Float16* featH = (_Float16*)alloc((size_t)FEATC * 2);
  _Float16* attnH = (_Float16*)alloc((size_t)FEATC * 2);
  float* x_all  = (float*)alloc((size_t)NALL * NFc * 4);
  float* dc_all = (float*)alloc((size_t)NALL * NFc * 4);
  float* aexp   = (float*)alloc((size_t)NALL * NFc * 4);
  float* sq     = (float*)alloc((size_t)NALL * 4);
  float* adj    = (float*)alloc((size_t)NALL * NALL * 4);
  int*   glab   = (int*)alloc((size_t)Bc * 4);
  float* xg     = (float*)alloc((size_t)NALL * HIDc * 4);
  float* gemb   = (float*)alloc((size_t)KCc * NFc * 4);
  float* gsc    = (float*)alloc((size_t)KCc * NFc * 4);
  _Float16* Hout = (_Float16*)alloc((size_t)Bc * Tc * NFc * 2);
  float* seqb   = (float*)alloc((size_t)Bc * Tc * HIDc * 4);

  k_micro_gru<<<dim3(NLc, Bc / 128), 128, 0, stream>>>(x, mc_w_ih, mc_w_hh, mc_b_ih, mc_b_hh, featH);
  k_static_proj<<<(Bc * FDc + 255) / 256, 256, 0, stream>>>(st, st_w, st_b, featH);
  k_attention<<<Bc, 256, 0, stream>>>(featH, td, wq, bq, wk, bk, wv, bv, decay, gfe,
                                      gm, conf_p, attnH, x_all, dc_all);
  k_centers<<<(KCc * NFc + 255) / 256, 256, 0, stream>>>(raw_c, cp_w, cp_b, cconf, x_all, dc_all);
  k_prep<<<(NALL + 255) / 256, 256, 0, stream>>>(x_all, dc_all, sq, aexp);
  k_adj<<<(NALL * NALL + 255) / 256, 256, 0, stream>>>(x_all, aexp, sq, adj_p, adj);
  k_glabel<<<(Bc + 255) / 256, 256, 0, stream>>>(adj, glab);
  k_xg<<<(NALL * HIDc + 255) / 256, 256, 0, stream>>>(x_all, gcn_w, xg);
  k_graph_small<<<1, 256, 0, stream>>>(adj, xg, gcn_b, eo_w, eo_b, imp1_w, imp1_b,
                                       imp2_w, imp2_b, gemb, gsc);
  k_p1<<<1280, 256, 0, stream>>>(attnH, p1_w, p1_b, gsc, gemb, glab, ge_p, Hout);
  k_p2<<<512, 256, 0, stream>>>(Hout, p2_w, p2_b, seqb);
  k_gru2<<<Bc / 256, 256, 0, stream>>>(seqb, gru_w_ih, gru_w_hh, gru_b_ih, gru_b_hh,
                                       head_w, head_b, (float*)d_out);
}
